// TransformerBlock_69904887710457
// MI455X (gfx1250) — matrix operand order repaired and checked by reading, not compiled, not run on hardware
//
#include <hip/hip_runtime.h>
#include <hip/hip_bf16.h>
#include <math.h>

#define DEV __device__ __forceinline__

typedef __attribute__((ext_vector_type(16))) __bf16 bf16x16;
typedef __attribute__((ext_vector_type(8)))  float  f32x8;

constexpr int NTOK   = 2048;
constexpr int EDIM   = 768;
constexpr int HEADS  = 8;
constexpr int DHEAD  = 96;
constexpr int FFDIM  = 3072;
constexpr int QKVDIM = 2304;

DEV __bf16 f2bf(float f) {
  union { float f; unsigned u; } x; x.f = f;
  unsigned u = x.u;
  unsigned r = (u + 0x7FFFu + ((u >> 16) & 1u)) >> 16;   // RNE truncate
  union { unsigned short s; __bf16 b; } y; y.s = (unsigned short)r;
  return y.b;
}
DEV float bf2f(__bf16 b) {
  union { unsigned short s; __bf16 b; } x; x.b = b;
  union { unsigned u; float f; } y; y.u = (unsigned)x.s << 16;
  return y.f;
}

// --- CDNA5 async copy global->LDS (ASYNCcnt), 16B per lane --------------------
DEV void async_b128(const __bf16* g, __bf16* l) {
  unsigned lds = (unsigned)(unsigned long long)l;   // low 32 bits = LDS offset
  asm volatile("global_load_async_to_lds_b128 %0, %1, off"
               :: "v"(lds), "v"(g) : "memory");
}
DEV void wait_async0() { asm volatile("s_wait_asynccnt 0x0" ::: "memory"); }

// ---------------------------------------------------------------------------
// LayerNorm: one 256-thread block per token; fp32 in, bf16 out
// ---------------------------------------------------------------------------
__global__ __launch_bounds__(256) void ln_kernel(const float* __restrict__ x,
                                                 const float* __restrict__ g,
                                                 const float* __restrict__ b,
                                                 __bf16* __restrict__ o) {
  const int n = blockIdx.x;
  const float* xr = x + (size_t)n * EDIM;
  float v[3];
  float s = 0.f, s2 = 0.f;
#pragma unroll
  for (int i = 0; i < 3; ++i) {
    v[i] = xr[threadIdx.x + i * 256];
    s += v[i]; s2 += v[i] * v[i];
  }
#pragma unroll
  for (int off = 16; off > 0; off >>= 1) {
    s  += __shfl_xor(s,  off, 32);
    s2 += __shfl_xor(s2, off, 32);
  }
  __shared__ float red0[8], red1[8], mv[2];
  const int wid = threadIdx.x >> 5, lane = threadIdx.x & 31;
  if (lane == 0) { red0[wid] = s; red1[wid] = s2; }
  __syncthreads();
  if (threadIdx.x == 0) {
    float t = 0.f, t2 = 0.f;
#pragma unroll
    for (int i = 0; i < 8; ++i) { t += red0[i]; t2 += red1[i]; }
    float mean = t * (1.f / EDIM);
    float var  = t2 * (1.f / EDIM) - mean * mean;
    mv[0] = mean; mv[1] = rsqrtf(var + 1e-5f);
  }
  __syncthreads();
  const float mean = mv[0], rstd = mv[1];
  __bf16* orow = o + (size_t)n * EDIM;
#pragma unroll
  for (int i = 0; i < 3; ++i) {
    int c = threadIdx.x + i * 256;
    orow[c] = f2bf((v[i] - mean) * rstd * g[c] + b[c]);
  }
}

// ---------------------------------------------------------------------------
// Tiled transpose + convert: W[K,Nc] fp32 -> Wt[Nc,K] bf16 (one z per layer)
// ---------------------------------------------------------------------------
__global__ __launch_bounds__(256) void transpose_convert(const float* __restrict__ W,
                                                         __bf16* __restrict__ Wt,
                                                         const int K, const int Nc) {
  __shared__ __bf16 tile[32][33];
  const size_t lo = (size_t)blockIdx.z * K * Nc;
  const int nb = blockIdx.x * 32, kb = blockIdx.y * 32;
#pragma unroll
  for (int j = threadIdx.y; j < 32; j += 8)
    tile[threadIdx.x][j] = f2bf(W[lo + (size_t)(kb + j) * Nc + nb + threadIdx.x]);
  __syncthreads();
#pragma unroll
  for (int j = threadIdx.y; j < 32; j += 8)
    Wt[lo + (size_t)(nb + j) * K + kb + threadIdx.x] = tile[j][threadIdx.x];
}

// ---------------------------------------------------------------------------
// GEMM: C[M,Nc] = A[M,K](bf16) @ Bt[Nc,K]^T(bf16) (+bias, +residual / +GELU)
// 256 threads = 8 waves; block tile 128x128, wave 32x64, K-step 32.
// Tiles staged with global_load_async_to_lds_b128, double-buffered (ASYNCcnt).
// EPI: 0 = +bias -> bf16 ; 1 = +bias +R(f32) -> f32 ; 2 = +bias +GELU -> bf16
// ---------------------------------------------------------------------------
template <int EPI>
__global__ __launch_bounds__(256) void gemm_kernel(const __bf16* __restrict__ A,
                                                   const __bf16* __restrict__ Bt,
                                                   const float* __restrict__ bias,
                                                   const float* __restrict__ R,
                                                   float* __restrict__ Cf,
                                                   __bf16* __restrict__ Cb,
                                                   const int K, const int Nc) {
  __shared__ __bf16 As[2][128][40];   // [m][k], 80B row stride (16B aligned, pad)
  __shared__ __bf16 Bs[2][128][40];   // [n][k]

  const int m0 = blockIdx.y * 128;
  const int n0 = blockIdx.x * 128;
  const int t    = threadIdx.x;
  const int lane = t & 31;
  const int wave = t >> 5;
  const int wm = wave >> 1;            // 0..3  (32 rows each)
  const int wn = wave & 1;             // 0..1  (64 cols each)
  const int ll = lane & 15;
  const int hi = lane >> 4;

  auto issue = [&](int bsel, int k0) {
#pragma unroll
    for (int l = 0; l < 2; ++l) {      // A tile: 128 rows x 64B = 512 chunks
      int c = t + l * 256, row = c >> 2, part = c & 3;
      async_b128(A + (size_t)(m0 + row) * K + k0 + part * 8, &As[bsel][row][part * 8]);
    }
#pragma unroll
    for (int l = 0; l < 2; ++l) {      // B tile: 128 rows x 64B
      int c = t + l * 256, row = c >> 2, part = c & 3;
      async_b128(Bt + (size_t)(n0 + row) * K + k0 + part * 8, &Bs[bsel][row][part * 8]);
    }
  };

  f32x8 acc[2][4];
#pragma unroll
  for (int i = 0; i < 2; ++i)
#pragma unroll
    for (int j = 0; j < 4; ++j)
#pragma unroll
      for (int e = 0; e < 8; ++e) acc[i][j][e] = 0.f;

  issue(0, 0);
  wait_async0();
  __syncthreads();

  int buf = 0;
  for (int k0 = 0; k0 < K; k0 += 32) {
    if (k0 + 32 < K) issue(buf ^ 1, k0 + 32);

    bf16x16 af[2], bfr[4];
#pragma unroll
    for (int mt = 0; mt < 2; ++mt) {   // A frag: 16x32 bf16 layout
      int row = wm * 32 + mt * 16 + ll;
#pragma unroll
      for (int i = 0; i < 8; ++i) {
        int ka = ((i < 4) ? 2 * i : 2 * i + 8) + hi * 8;
        af[mt][2 * i]     = As[buf][row][ka];
        af[mt][2 * i + 1] = As[buf][row][ka + 1];
      }
    }
#pragma unroll
    for (int nt = 0; nt < 4; ++nt) {   // B frag: 32x16 bf16 layout
      int col = wn * 64 + nt * 16 + ll;
#pragma unroll
      for (int i = 0; i < 8; ++i) {
        int kb = ((i < 4) ? 2 * i : 2 * i + 8) + hi * 8;
        bfr[nt][2 * i]     = Bs[buf][col][kb];
        bfr[nt][2 * i + 1] = Bs[buf][col][kb + 1];
      }
    }
#pragma unroll
    for (int mt = 0; mt < 2; ++mt)
#pragma unroll
      for (int nt = 0; nt < 4; ++nt)
        acc[mt][nt] = __builtin_amdgcn_wmma_f32_16x16x32_bf16(
            false, af[mt], false, bfr[nt], (short)0, acc[mt][nt], false, false);

    wait_async0();
    __syncthreads();
    buf ^= 1;
  }

#pragma unroll
  for (int mt = 0; mt < 2; ++mt)
#pragma unroll
    for (int nt = 0; nt < 4; ++nt)
#pragma unroll
      for (int r = 0; r < 8; ++r) {
        int row = m0 + wm * 32 + mt * 16 + hi * 8 + r;
        int col = n0 + wn * 64 + nt * 16 + ll;
        float v = acc[mt][nt][r] + bias[col];
        size_t idx = (size_t)row * Nc + col;
        if (EPI == 2) v = 0.5f * v * (1.f + erff(v * 0.70710678118654752f));
        if (EPI == 1) Cf[idx] = v + R[idx];
        else          Cb[idx] = f2bf(v);
      }
}

// ---------------------------------------------------------------------------
// Split QKV [N,(h,d,3)] bf16 -> per-head bf16 Q/K/V [H][N][96]; Q pre-scaled
// ---------------------------------------------------------------------------
__global__ __launch_bounds__(256) void qkv_split_kernel(const __bf16* __restrict__ qkv,
                                                        __bf16* __restrict__ Q,
                                                        __bf16* __restrict__ Kd,
                                                        __bf16* __restrict__ V) {
  int i = blockIdx.x * 256 + threadIdx.x;        // 0 .. NTOK*EDIM-1
  int n = i / EDIM;
  int hd = i - n * EDIM;
  int h = hd / DHEAD;
  int d = hd - h * DHEAD;
  const __bf16* src = qkv + (size_t)n * QKVDIM + hd * 3;
  size_t dst = ((size_t)h * NTOK + n) * DHEAD + d;
  const float qscale = 0.03608439182435161f;     // 1/sqrt(768) (energy/scale)
  Q[dst]  = f2bf(bf2f(src[0]) * qscale);
  Kd[dst] = src[1];
  V[dst]  = src[2];
}

// ---------------------------------------------------------------------------
// Flash attention: grid (N/64, H), 128 threads (4 waves).
// Q + K tiles via async-to-LDS (K double-buffered); V staged through uint4
// registers and scatter-transposed after the WMMA block. Online softmax.
// ---------------------------------------------------------------------------
__global__ __launch_bounds__(128) void attn_kernel(const __bf16* __restrict__ Q,
                                                   const __bf16* __restrict__ Kg,
                                                   const __bf16* __restrict__ Vg,
                                                   __bf16* __restrict__ O) {
  constexpr int BQ = 64, BK = 64, NB = NTOK / BK;
  __shared__ __bf16 sQ[BQ][104];              // 208B rows (16B aligned)
  __shared__ __bf16 sK[2][BK][104];
  __shared__ __bf16 sVt[2][DHEAD][72];        // transposed: key pairs contiguous
  __shared__ __bf16 sP[BQ][72];

  const int q0 = blockIdx.x * BQ;
  const int h  = blockIdx.y;
  const int t    = threadIdx.x;
  const int lane = t & 31;
  const int w  = t >> 5;                      // 0..3
  const int ll = lane & 15;
  const int hi = lane >> 4;
  const int vrow = t >> 1, vpart = t & 1;     // V staging: 2 threads per row

  auto issueK = [&](int bsel, int kb) {       // 64 rows x 192B = 768 chunks
    const __bf16* src = Kg + ((size_t)h * NTOK + kb * BK) * DHEAD;
#pragma unroll
    for (int l = 0; l < 6; ++l) {
      int c = t + l * 128, row = c / 12, part = c % 12;
      async_b128(src + (size_t)row * DHEAD + part * 8, &sK[bsel][row][part * 8]);
    }
  };

  {                                           // async Q block 64x96
    const __bf16* src = Q + ((size_t)h * NTOK + q0) * DHEAD;
#pragma unroll
    for (int l = 0; l < 6; ++l) {
      int c = t + l * 128, row = c / 12, part = c % 12;
      async_b128(src + (size_t)row * DHEAD + part * 8, &sQ[row][part * 8]);
    }
  }

  uint4 vr[6];
  auto loadV = [&](int kb) {
    const __bf16* vs = Vg + ((size_t)h * NTOK + kb * BK) * DHEAD;
#pragma unroll
    for (int j = 0; j < 6; ++j)
      vr[j] = *(const uint4*)(vs + (size_t)vrow * DHEAD + vpart * 48 + j * 8);
  };
  auto storeV = [&](int bsel) {
#pragma unroll
    for (int j = 0; j < 6; ++j) {
      union { uint4 u; __bf16 e[8]; } x; x.u = vr[j];
#pragma unroll
      for (int k = 0; k < 8; ++k)
        sVt[bsel][vpart * 48 + j * 8 + k][vrow] = x.e[k];
    }
  };

  f32x8 oacc[6];
#pragma unroll
  for (int nt = 0; nt < 6; ++nt)
#pragma unroll
    for (int e = 0; e < 8; ++e) oacc[nt][e] = 0.f;
  float mrun[8], lrun[8];
#pragma unroll
  for (int r = 0; r < 8; ++r) { mrun[r] = -3.0e38f; lrun[r] = 0.f; }

  issueK(0, 0);
  loadV(0);
  storeV(0);
  wait_async0();
  __syncthreads();

  int buf = 0;
  for (int kb = 0; kb < NB; ++kb) {
    const bool more = (kb + 1) < NB;
    if (more) { issueK(buf ^ 1, kb + 1); loadV(kb + 1); }

    // --- S = Q @ K^T (16 q-rows x 64 keys per wave) ---
    f32x8 sacc[4];
#pragma unroll
    for (int nt = 0; nt < 4; ++nt)
#pragma unroll
      for (int e = 0; e < 8; ++e) sacc[nt][e] = 0.f;

#pragma unroll
    for (int kc = 0; kc < 3; ++kc) {          // dh = 96 = 3 * 32
      const int d0 = kc * 32;
      bf16x16 af;
      int row = w * 16 + ll;
#pragma unroll
      for (int i = 0; i < 8; ++i) {
        int ka = d0 + ((i < 4) ? 2 * i : 2 * i + 8) + hi * 8;
        af[2 * i]     = sQ[row][ka];
        af[2 * i + 1] = sQ[row][ka + 1];
      }
#pragma unroll
      for (int nt = 0; nt < 4; ++nt) {
        bf16x16 bfr;
        int key = nt * 16 + ll;
#pragma unroll
        for (int i = 0; i < 8; ++i) {
          int kbp = d0 + ((i < 4) ? 2 * i : 2 * i + 8) + hi * 8;
          bfr[2 * i]     = sK[buf][key][kbp];
          bfr[2 * i + 1] = sK[buf][key][kbp + 1];
        }
        sacc[nt] = __builtin_amdgcn_wmma_f32_16x16x32_bf16(
            false, af, false, bfr, (short)0, sacc[nt], false, false);
      }
    }

    // --- online softmax (row = hi*8 + r lives in one 16-lane half) ---
#pragma unroll
    for (int r = 0; r < 8; ++r) {
      float mx = sacc[0][r];
#pragma unroll
      for (int nt = 1; nt < 4; ++nt) mx = fmaxf(mx, sacc[nt][r]);
#pragma unroll
      for (int off = 8; off > 0; off >>= 1) mx = fmaxf(mx, __shfl_xor(mx, off, 32));
      float mnew = fmaxf(mrun[r], mx);
      float corr = __expf(mrun[r] - mnew);
      float rs = 0.f;
#pragma unroll
      for (int nt = 0; nt < 4; ++nt) {
        float p = __expf(sacc[nt][r] - mnew);
        sacc[nt][r] = p;
        rs += p;
      }
#pragma unroll
      for (int off = 8; off > 0; off >>= 1) rs += __shfl_xor(rs, off, 32);
      lrun[r] = lrun[r] * corr + rs;
      mrun[r] = mnew;
#pragma unroll
      for (int nt = 0; nt < 6; ++nt) oacc[nt][r] *= corr;
    }

    // --- P (D-layout) -> LDS -> A-layout for P @ V (wave-private rows) ---
#pragma unroll
    for (int nt = 0; nt < 4; ++nt)
#pragma unroll
      for (int r = 0; r < 8; ++r)
        sP[w * 16 + hi * 8 + r][nt * 16 + ll] = f2bf(sacc[nt][r]);

#pragma unroll
    for (int kk = 0; kk < 2; ++kk) {          // 64 keys = 2 * 32
      bf16x16 pf;
      int row = w * 16 + ll;
#pragma unroll
      for (int i = 0; i < 8; ++i) {
        int ka = kk * 32 + ((i < 4) ? 2 * i : 2 * i + 8) + hi * 8;
        pf[2 * i]     = sP[row][ka];
        pf[2 * i + 1] = sP[row][ka + 1];
      }
#pragma unroll
      for (int nt = 0; nt < 6; ++nt) {
        bf16x16 vf;
        int d = nt * 16 + ll;
#pragma unroll
        for (int i = 0; i < 8; ++i) {
          int kkey = kk * 32 + ((i < 4) ? 2 * i : 2 * i + 8) + hi * 8;
          vf[2 * i]     = sVt[buf][d][kkey];
          vf[2 * i + 1] = sVt[buf][d][kkey + 1];
        }
        oacc[nt] = __builtin_amdgcn_wmma_f32_16x16x32_bf16(
            false, pf, false, vf, (short)0, oacc[nt], false, false);
      }
    }

    if (more) storeV(buf ^ 1);    // ds stores of next V overlap past compute
    wait_async0();
    __syncthreads();
    buf ^= 1;
  }

  // --- normalize and write O (bf16) in [n, h*96+d] layout ---
#pragma unroll
  for (int nt = 0; nt < 6; ++nt)
#pragma unroll
    for (int r = 0; r < 8; ++r) {
      int row = q0 + w * 16 + hi * 8 + r;
      int col = h * DHEAD + nt * 16 + ll;
      O[(size_t)row * EDIM + col] = f2bf(oacc[nt][r] / lrun[r]);
    }
}

// ---------------------------------------------------------------------------
// Host orchestration
// ---------------------------------------------------------------------------
extern "C" void kernel_launch(void* const* d_in, const int* in_sizes, int n_in,
                              void* d_out, int out_size, void* d_ws, size_t ws_size,
                              hipStream_t stream) {
  (void)in_sizes; (void)n_in; (void)out_size; (void)ws_size;
  const float* x_in   = (const float*)d_in[0];
  const float* ln1_g  = (const float*)d_in[1];
  const float* ln1_b  = (const float*)d_in[2];
  const float* qkv_w  = (const float*)d_in[3];
  const float* qkv_b  = (const float*)d_in[4];
  const float* proj_w = (const float*)d_in[5];
  const float* proj_b = (const float*)d_in[6];
  const float* ln2_g  = (const float*)d_in[7];
  const float* ln2_b  = (const float*)d_in[8];
  const float* ff1_w  = (const float*)d_in[9];
  const float* ff1_b  = (const float*)d_in[10];
  const float* ff2_w  = (const float*)d_in[11];
  const float* ff2_b  = (const float*)d_in[12];
  float* out = (float*)d_out;

  char* ws = (char*)d_ws;
  size_t off = 0;
  auto alloc = [&](size_t bytes) {
    char* p = ws + off;
    off += (bytes + 255) & ~(size_t)255;
    return p;
  };
  float*  xcur    = (float*) alloc((size_t)NTOK * EDIM * 4);
  __bf16* hbuf    = (__bf16*)alloc((size_t)NTOK * EDIM * 2);
  __bf16* qkvbuf  = (__bf16*)alloc((size_t)NTOK * QKVDIM * 2);
  __bf16* attno   = (__bf16*)alloc((size_t)NTOK * EDIM * 2);
  __bf16* ffbuf   = (__bf16*)alloc((size_t)NTOK * FFDIM * 2);
  __bf16* Qh      = (__bf16*)alloc((size_t)HEADS * NTOK * DHEAD * 2);
  __bf16* Kh      = (__bf16*)alloc((size_t)HEADS * NTOK * DHEAD * 2);
  __bf16* Vh      = (__bf16*)alloc((size_t)HEADS * NTOK * DHEAD * 2);
  __bf16* qkv_wt  = (__bf16*)alloc((size_t)12 * EDIM * QKVDIM * 2);  // [Nc,K]
  __bf16* proj_wt = (__bf16*)alloc((size_t)12 * EDIM * EDIM * 2);
  __bf16* ff1_wt  = (__bf16*)alloc((size_t)12 * EDIM * FFDIM * 2);
  __bf16* ff2_wt  = (__bf16*)alloc((size_t)12 * FFDIM * EDIM * 2);

  // one-time (per launch) weight convert+transpose to bf16 [Nc,K]
  dim3 tb(32, 8);
  transpose_convert<<<dim3(QKVDIM / 32, EDIM / 32, 12), tb, 0, stream>>>(qkv_w,  qkv_wt,  EDIM,  QKVDIM);
  transpose_convert<<<dim3(EDIM  / 32, EDIM / 32, 12), tb, 0, stream>>>(proj_w, proj_wt, EDIM,  EDIM);
  transpose_convert<<<dim3(FFDIM / 32, EDIM / 32, 12), tb, 0, stream>>>(ff1_w,  ff1_wt,  EDIM,  FFDIM);
  transpose_convert<<<dim3(EDIM  / 32, FFDIM / 32, 12), tb, 0, stream>>>(ff2_w,  ff2_wt,  FFDIM, EDIM);

  hipMemcpyAsync(xcur, x_in, (size_t)NTOK * EDIM * 4, hipMemcpyDeviceToDevice, stream);

  int outIdx = 0;
  for (int l = 0; l < 12; ++l) {
    ln_kernel<<<NTOK, 256, 0, stream>>>(xcur, ln1_g + l * EDIM, ln1_b + l * EDIM, hbuf);
    gemm_kernel<0><<<dim3(QKVDIM / 128, NTOK / 128), 256, 0, stream>>>(
        hbuf, qkv_wt + (size_t)l * EDIM * QKVDIM, qkv_b + (size_t)l * QKVDIM,
        nullptr, nullptr, qkvbuf, EDIM, QKVDIM);
    qkv_split_kernel<<<(NTOK * EDIM) / 256, 256, 0, stream>>>(qkvbuf, Qh, Kh, Vh);
    attn_kernel<<<dim3(NTOK / 64, HEADS), 128, 0, stream>>>(Qh, Kh, Vh, attno);
    gemm_kernel<1><<<dim3(EDIM / 128, NTOK / 128), 256, 0, stream>>>(
        attno, proj_wt + (size_t)l * EDIM * EDIM, proj_b + (size_t)l * EDIM,
        xcur, xcur, nullptr, EDIM, EDIM);
    ln_kernel<<<NTOK, 256, 0, stream>>>(xcur, ln2_g + l * EDIM, ln2_b + l * EDIM, hbuf);
    gemm_kernel<2><<<dim3(FFDIM / 128, NTOK / 128), 256, 0, stream>>>(
        hbuf, ff1_wt + (size_t)l * EDIM * FFDIM, ff1_b + (size_t)l * FFDIM,
        nullptr, nullptr, ffbuf, EDIM, FFDIM);
    gemm_kernel<1><<<dim3(EDIM / 128, NTOK / 128), 256, 0, stream>>>(
        ffbuf, ff2_wt + (size_t)l * FFDIM * EDIM, ff2_b + (size_t)l * EDIM,
        xcur, xcur, nullptr, FFDIM, EDIM);
    if (l == 2 || l == 5 || l == 8 || l == 11) {
      hipMemcpyAsync(out + (size_t)outIdx * NTOK * EDIM, xcur,
                     (size_t)NTOK * EDIM * 4, hipMemcpyDeviceToDevice, stream);
      ++outIdx;
    }
  }
}
